// MyMamba_82016695484463
// MI455X (gfx1250) — matrix operand order repaired and checked by reading, not compiled, not run on hardware
//
#include <hip/hip_runtime.h>

// ---------------------------------------------------------------------------
// Problem constants (from reference)
// ---------------------------------------------------------------------------
#define Bc     128
#define NPc    64
#define Pc     128
#define DIMc   512
#define DSTc   16
#define DCONVc 4
#define DEPTHc 2
#define DINc   1024      // 2*DIM
#define DTRc   32        // DIM/16
#define NCLSc  32
#define ROWS   (Bc * NPc)   // 8192 flattened (b, l) rows

typedef __attribute__((ext_vector_type(16))) __bf16    v16bf;
typedef __attribute__((ext_vector_type(8)))  __bf16    v8bf;
typedef __attribute__((ext_vector_type(8)))  float     v8f;
typedef __attribute__((ext_vector_type(4)))  unsigned  v4u;
typedef __attribute__((ext_vector_type(8)))  int       v8i;
typedef __attribute__((ext_vector_type(4)))  int       v4i;

#if __has_builtin(__builtin_amdgcn_tensor_load_to_lds)
#define HAS_TDM 1
#else
#define HAS_TDM 0
#endif

// ---------------------------------------------------------------------------
// Helpers
// ---------------------------------------------------------------------------
__device__ __forceinline__ __bf16 f2bf(float f) {
    union { float f; unsigned u; } x; x.f = f;
    unsigned r = x.u + 0x7FFFu + ((x.u >> 16) & 1u);   // round-to-nearest-even
    unsigned short h = (unsigned short)(r >> 16);
    __bf16 out;
    __builtin_memcpy(&out, &h, 2);
    return out;
}

__device__ __forceinline__ float softplus_f(float x) {
    return x > 20.0f ? x : log1pf(__expf(x));
}

__device__ __forceinline__ void wait_tensorcnt0() {
#if __has_builtin(__builtin_amdgcn_s_wait_tensorcnt)
    __builtin_amdgcn_s_wait_tensorcnt(0);
#else
    asm volatile("s_wait_tensorcnt 0x0" ::: "memory");
#endif
}
__device__ __forceinline__ void wait_tensorcnt1() {
#if __has_builtin(__builtin_amdgcn_s_wait_tensorcnt)
    __builtin_amdgcn_s_wait_tensorcnt(1);
#else
    asm volatile("s_wait_tensorcnt 0x1" ::: "memory");
#endif
}

#define SHUF16(lo, hi) __builtin_shufflevector(lo, hi, \
    0, 1, 2, 3, 4, 5, 6, 7, 8, 9, 10, 11, 12, 13, 14, 15)

#define WT_STRIDE 40   // LDS row stride (elements): 64B data + 16B pad = 80B
#define WT_BUFBYTES (64 * WT_STRIDE * 2)   // bytes per double-buffer slot

#if HAS_TDM
// ---------------------------------------------------------------------------
// TDM: DMA a [rows x 32] bf16 tile (row stride ldw elements) from global into
// LDS at byte offset ldsAddr, padding +16B after each 64B row (-> 80B stride,
// conflict-free for b128 B-fragment reads). D# per CDNA5 ISA.
// ---------------------------------------------------------------------------
__device__ __forceinline__ void tdm_load_tile(const __bf16* gsrc, unsigned ldsAddr,
                                              int rows, int ldw) {
    const unsigned long long ga = (unsigned long long)(uintptr_t)gsrc;
    v4u g0;
    g0.x = 1u;                                           // count=1, user mode
    g0.y = ldsAddr;                                      // lds_addr (bytes)
    g0.z = (unsigned)(ga & 0xFFFFFFFFull);               // global_addr[31:0]
    g0.w = (unsigned)((ga >> 32) & 0x1FFFFFFull)         // global_addr[56:32]
         | (2u << 30);                                   // type = 2 ("image")
    v8i g1;
    g1[0] = (1 << 16)        // data_size = 1 -> 2 bytes
          | (1 << 20)        // pad_enable
          | (3 << 22)        // pad_interval: 16 DWORDs (= 64B row)
          | (3 << 25);       // pad_amount: 4 DWORDs (= 16B)
    g1[1] = (int)(32u << 16);            // tensor_dim0 = 32 (bits 79:48 lo)
    g1[2] = (int)((unsigned)rows << 16); // tensor_dim0 hi=0, tensor_dim1 lo
    g1[3] = (int)(32u << 16);            // tensor_dim1 hi=0, tile_dim0 = 32
    g1[4] = rows;                        // tile_dim1 = rows, tile_dim2 = 0
    g1[5] = ldw;                         // tensor_dim0_stride (elements)
    g1[6] = 0;
    g1[7] = 0;
    v4i z4 = {};
#if __clang_major__ >= 23
    v8i z8 = {};
    __builtin_amdgcn_tensor_load_to_lds(g0, g1, z4, z4, z8, 0);
#else
    __builtin_amdgcn_tensor_load_to_lds(g0, g1, z4, z4, 0);
#endif
}
#endif  // HAS_TDM

// ---------------------------------------------------------------------------
// bf16 WMMA GEMM:  C[M, NT*16*gridX] (+)= A[M,K] * W^T,  A,W bf16, W:[N,K].
//   blockDim = 128 (4 waves). Block tile 64(M) x NT*16(N); wave tile 16 x NT*16.
//   W tile double-buffered in LDS via TDM (fallback: VGPR staging).
//   A fragments register-prefetched. NT compile-time -> branchless K-loop.
//   ACT: 0 = none, 1 = softplus.  Optional bf16 mirror of C (Cbf).
//   blockIdx.z batches GEMMs / split-K via element strides aSz/wSz/cSz/bSz.
//   Requires: M % 64 == 0, K % 32 == 0, lda/ldw % 8 == 0 (16B-aligned rows).
// ---------------------------------------------------------------------------
template <int ACT, int NT>
__global__ void gemm_bf16_wmma_kernel(const __bf16* __restrict__ A, long lda, long aSz,
                                      const __bf16* __restrict__ W, long ldw, long wSz,
                                      float* __restrict__ C, long ldc, long cSz,
                                      const float* __restrict__ bias, long bSz,
                                      __bf16* __restrict__ Cbf, long ldcb,
                                      int K, int atomicAccum) {
    __shared__ __bf16 Wtile[2][64 * WT_STRIDE];

    const int tid  = threadIdx.x;
    const int lane = tid & 31;
    const int wave = tid >> 5;
    const long z  = blockIdx.z;
    A += z * aSz;
    W += z * wSz;
    C += z * cSz;
    if (bias) bias += z * bSz;

    const int mBase  = blockIdx.y * 64 + wave * 16;
    const int nBase0 = blockIdx.x * (NT * 16);

    // A-fragment lane mapping (ISA 7.12.2, 16-bit A 16x32)
    const int am  = lane & 15;
    const int akg = (lane >> 4) << 3;    // 0 or 8
    // B-fragment lane mapping (32x16, row striped over lanes)
    const int bn  = lane & 15;
    const int bko = (lane >> 4) << 3;    // 0 or 8

    const __bf16* Arow  = A + (long)(mBase + am) * lda;
    const __bf16* Wtile0 = W + (long)nBase0 * ldw;

#if !HAS_TDM
    const int srow  = tid >> 1;
    const int skoff = (tid & 1) << 4;
    const bool sAct = srow < NT * 16;
    const __bf16* Wst = Wtile0 + (long)srow * ldw + skoff;
#endif

    v8f acc[NT] = {};

    // ---- prologue: stage tile 0, preload A fragment 0 ----
#if HAS_TDM
    if (wave == 0) tdm_load_tile(Wtile0, 0u, NT * 16, (int)ldw);
#else
    {
        uint4 wv = {0u, 0u, 0u, 0u};
        if (sAct) wv = *(const uint4*)(Wst);
        *(uint4*)(&Wtile[0][srow * WT_STRIDE + skoff]) = wv;
    }
#endif
    v8bf alo = *(const v8bf*)(Arow + akg);
    v8bf ahi = *(const v8bf*)(Arow + 16 + akg);

    int cur = 0;
    for (int k = 0; k < K; k += 32) {
        const bool more = (k + 32) < K;

#if HAS_TDM
        __syncthreads();              // everyone done reading Wtile[cur^1]
        if (wave == 0) {
            if (more) tdm_load_tile(Wtile0 + k + 32,
                                    (unsigned)((cur ^ 1) * WT_BUFBYTES),
                                    NT * 16, (int)ldw);
            if (more) wait_tensorcnt1();   // tile 'cur' done; next may fly
            else      wait_tensorcnt0();
        }
        __syncthreads();              // Wtile[cur] visible to all waves
        v8bf nlo = {}, nhi = {};
        if (more) {                   // register-prefetch next A fragment
            nlo = *(const v8bf*)(Arow + k + 32 + akg);
            nhi = *(const v8bf*)(Arow + k + 48 + akg);
        }
#else
        __syncthreads();              // Wtile[cur] ready for all waves
        uint4 wnext = {0u, 0u, 0u, 0u};
        v8bf nlo = {}, nhi = {};
        if (more) {                   // issue next-tile loads early
            if (sAct) wnext = *(const uint4*)(Wst + k + 32);
            nlo = *(const v8bf*)(Arow + k + 32 + akg);
            nhi = *(const v8bf*)(Arow + k + 48 + akg);
        }
#endif

        const v16bf afrag = SHUF16(alo, ahi);

        // load all NT B fragments first (one DS clause, single wait) ...
        v16bf bfrag[NT];
#pragma unroll
        for (int t = 0; t < NT; ++t) {
            const __bf16* wp = &Wtile[cur][(t * 16 + bn) * WT_STRIDE + bko];
            v8bf blo = *(const v8bf*)(wp);
            v8bf bhi = *(const v8bf*)(wp + 16);
            bfrag[t] = SHUF16(blo, bhi);
        }
        // ... then issue the WMMAs back-to-back
#pragma unroll
        for (int t = 0; t < NT; ++t)
            acc[t] = __builtin_amdgcn_wmma_f32_16x16x32_bf16(
                false, afrag, false, bfrag[t], (short)0, acc[t], false, false);

#if !HAS_TDM
        if (more) {                   // fill the other buffer
            *(uint4*)(&Wtile[cur ^ 1][srow * WT_STRIDE + skoff]) = wnext;
        }
#endif
        if (more) { alo = nlo; ahi = nhi; }
        cur ^= 1;
    }

    // ---- epilogue: C/D layout N = lane&15, M = (lane>>4)*8 + vgpr ----
    const int n    = lane & 15;
    const int mOff = (lane >> 4) << 3;
#pragma unroll
    for (int t = 0; t < NT; ++t) {
        const int nBase = nBase0 + t * 16;
        const float bv = bias ? bias[nBase + n] : 0.0f;
#pragma unroll
        for (int r = 0; r < 8; ++r) {
            float val = acc[t][r] + bv;
            if (ACT == 1) val = softplus_f(val);
            const long mrow = mBase + mOff + r;
            float* dst = &C[mrow * ldc + nBase + n];
            if (atomicAccum) atomicAdd(dst, val);
            else             *dst = val;
            if (Cbf) Cbf[mrow * ldcb + nBase + n] = f2bf(val);
        }
    }
}

// ---------------------------------------------------------------------------
// fp32 -> bf16 converter (grid-stride)
// ---------------------------------------------------------------------------
__global__ void cvt_bf16_kernel(const float* __restrict__ in, __bf16* __restrict__ out,
                                long n) {
    for (long i = (long)blockIdx.x * 256 + threadIdx.x; i < n; i += (long)gridDim.x * 256)
        out[i] = f2bf(in[i]);
}

// seg_w [NP][P][DIM] -> bf16 transposed [NP][DIM][P]
__global__ void seg_transpose_kernel(const float* __restrict__ in, __bf16* __restrict__ out) {
    const long idx = (long)blockIdx.x * 256 + threadIdx.x;  // NP*DIM*P
    const int p  = (int)(idx & (Pc - 1));
    const int d  = (int)((idx >> 7) & (DIMc - 1));
    const int nn = (int)(idx >> 16);
    out[idx] = f2bf(in[(long)nn * Pc * DIMc + (long)p * DIMc + d]);
}

// ---------------------------------------------------------------------------
// LayerNorm over last dim D, one block (256 threads) per row; optional bf16 out.
// ---------------------------------------------------------------------------
__global__ void layernorm_kernel(const float* __restrict__ X, float* __restrict__ Y,
                                 __bf16* __restrict__ Ybf,
                                 const float* __restrict__ g, const float* __restrict__ bta,
                                 int D, int gRows) {
    const long row = blockIdx.x;
    const float* x = X + row * (long)D;
    float* y = Y + row * (long)D;
    __bf16* yb = Ybf ? Ybf + row * (long)D : nullptr;
    const long grow = (gRows > 1) ? (row % gRows) : 0;
    const float* gp = g + grow * (long)D;
    const float* bp = bta + grow * (long)D;

    float s = 0.0f, s2 = 0.0f;
    for (int i = threadIdx.x; i < D; i += 256) {
        float v = x[i];
        s += v; s2 += v * v;
    }
#pragma unroll
    for (int off = 16; off > 0; off >>= 1) {
        s  += __shfl_down(s, off, 32);
        s2 += __shfl_down(s2, off, 32);
    }
    __shared__ float red[2][8];
    __shared__ float mr[2];
    const int lane = threadIdx.x & 31, wv = threadIdx.x >> 5;
    if (lane == 0) { red[0][wv] = s; red[1][wv] = s2; }
    __syncthreads();
    if (threadIdx.x == 0) {
        float ts = 0.0f, ts2 = 0.0f;
#pragma unroll
        for (int w = 0; w < 8; ++w) { ts += red[0][w]; ts2 += red[1][w]; }
        float mu  = ts / (float)D;
        float var = ts2 / (float)D - mu * mu;
        mr[0] = mu;
        mr[1] = rsqrtf(var + 1e-5f);
    }
    __syncthreads();
    const float mu = mr[0], rstd = mr[1];
    for (int i = threadIdx.x; i < D; i += 256) {
        float v = (x[i] - mu) * rstd * gp[i] + bp[i];
        y[i] = v;
        if (yb) yb[i] = f2bf(v);
    }
}

// ---------------------------------------------------------------------------
// Depthwise causal conv1d (kernel 4, pad 3) + SiLU. Writes f32 + bf16 mirrors.
// ---------------------------------------------------------------------------
__global__ void conv_silu_kernel(const float* __restrict__ xz,
                                 const float* __restrict__ cw,   // [DIN,4]
                                 const float* __restrict__ cb,   // [DIN]
                                 float* __restrict__ uc,         // [ROWS,DIN]
                                 __bf16* __restrict__ ucbf) {
    const long idx = (long)blockIdx.x * 256 + threadIdx.x;       // ROWS*DIN
    const int  e  = (int)(idx & (DINc - 1));
    const long r  = idx >> 10;
    const int  nn = (int)(r & (NPc - 1));
    const long rb = r - nn;                                      // l = 0 row
    float acc = cb[e];
#pragma unroll
    for (int k = 0; k < DCONVc; ++k) {
        const int l = nn - (DCONVc - 1) + k;
        if (l >= 0) acc += xz[(rb + l) * (2 * DINc) + e] * cw[e * DCONVc + k];
    }
    const float sig = 1.0f / (1.0f + __expf(-acc));
    const float v = acc * sig;
    uc[idx]   = v;
    ucbf[idx] = f2bf(v);
}

// ---------------------------------------------------------------------------
// Selective scan. Block (256 threads) = one batch b, 256 channels.
// h[16] in registers; B_t/C_t broadcast via LDS. Fused D-skip + SiLU gate.
// Output written directly as bf16 (only consumer is out_proj WMMA).
// ---------------------------------------------------------------------------
__global__ void scan_kernel(const float* __restrict__ delta,  // [ROWS,DIN]
                            const float* __restrict__ uc,     // [ROWS,DIN]
                            const float* __restrict__ xz,     // [ROWS,2*DIN] (z half)
                            const float* __restrict__ xdbl,   // [ROWS,64] (B@32, C@48)
                            const float* __restrict__ A_log,  // [DIN,16]
                            const float* __restrict__ Dp,     // [DIN]
                            __bf16* __restrict__ y) {         // [ROWS,DIN]
    const int b = blockIdx.y;
    const int e = blockIdx.x * 256 + threadIdx.x;

    float Arow[DSTc];
#pragma unroll
    for (int s = 0; s < DSTc; ++s) Arow[s] = -__expf(A_log[e * DSTc + s]);
    const float Dv = Dp[e];

    float h[DSTc];
#pragma unroll
    for (int s = 0; s < DSTc; ++s) h[s] = 0.0f;

    __shared__ float BC[2 * DSTc];

    for (int l = 0; l < NPc; ++l) {
        const long r = (long)b * NPc + l;
        if (threadIdx.x < 2 * DSTc)
            BC[threadIdx.x] = xdbl[r * 64 + DTRc + threadIdx.x];
        __syncthreads();

        const float dt = delta[r * DINc + e];
        const float u  = uc[r * DINc + e];
        const float du = dt * u;
        float ys = 0.0f;
#pragma unroll
        for (int s = 0; s < DSTc; ++s) {
            const float dA = __expf(dt * Arow[s]);
            h[s] = dA * h[s] + du * BC[s];
            ys += h[s] * BC[DSTc + s];
        }
        const float zv  = xz[r * (2 * DINc) + DINc + e];
        const float sil = zv / (1.0f + __expf(-zv));
        y[r * DINc + e] = f2bf((ys + u * Dv) * sil);
        __syncthreads();
    }
}

// ---------------------------------------------------------------------------
// Seed classifier output with bias (split-K head GEMM accumulates on top).
// ---------------------------------------------------------------------------
__global__ void init_out_kernel(float* __restrict__ out, const float* __restrict__ hb) {
    const int i = blockIdx.x * 256 + threadIdx.x;
    if (i < Bc * NCLSc) out[i] = hb[i & (NCLSc - 1)];
}

// ---------------------------------------------------------------------------
// Host orchestration
// ---------------------------------------------------------------------------
extern "C" void kernel_launch(void* const* d_in, const int* in_sizes, int n_in,
                              void* d_out, int out_size, void* d_ws, size_t ws_size,
                              hipStream_t stream) {
    const float* series    = (const float*)d_in[0];
    const float* seg_w     = (const float*)d_in[1];
    const float* seg_b     = (const float*)d_in[2];
    const float* ln_g      = (const float*)d_in[3];
    const float* ln_b      = (const float*)d_in[4];
    const float* in_proj_w = (const float*)d_in[5];
    const float* conv_w    = (const float*)d_in[6];
    const float* conv_b    = (const float*)d_in[7];
    const float* x_proj_w  = (const float*)d_in[8];
    const float* dt_proj_w = (const float*)d_in[9];
    const float* dt_proj_b = (const float*)d_in[10];
    const float* A_log     = (const float*)d_in[11];
    const float* Dmat      = (const float*)d_in[12];
    const float* out_proj_w= (const float*)d_in[13];
    const float* head_ln_g = (const float*)d_in[14];
    const float* head_ln_b = (const float*)d_in[15];
    const float* head_w    = (const float*)d_in[16];
    const float* head_b    = (const float*)d_in[17];
    float* out = (float*)d_out;

    // ---- workspace layout ----
    float* fws   = (float*)d_ws;
    float* x     = fws;                             // [8192, 512]
    float* xz    = x     + (long)ROWS * DIMc;       // [8192, 2048]
    float* uc    = xz    + (long)ROWS * 2 * DINc;   // [8192, 1024]
    float* xdbl  = uc    + (long)ROWS * DINc;       // [8192, 64]
    float* delta = xdbl  + (long)ROWS * 64;         // [8192, 1024]

    __bf16* bws       = (__bf16*)(delta + (long)ROWS * DINc);
    __bf16* series_bf = bws;                                      // 1,048,576
    __bf16* x_bf      = series_bf + (long)Bc * NPc * Pc;          // 4,194,304
    __bf16* uc_bf     = x_bf      + (long)ROWS * DIMc;            // 8,388,608
    __bf16* xdbl_bf   = uc_bf     + (long)ROWS * DINc;            // 524,288
    __bf16* y_bf      = xdbl_bf   + (long)ROWS * 64;              // 8,388,608
    __bf16* wseg_bf   = y_bf      + (long)ROWS * DINc;            // 4,194,304
    __bf16* win_bf    = wseg_bf   + (long)NPc * DIMc * Pc;        // 2,097,152
    __bf16* wx_bf     = win_bf    + (long)DEPTHc * 2 * DINc * DIMc;
    __bf16* wdt_bf    = wx_bf     + (long)DEPTHc * 64 * DINc;
    __bf16* wout_bf   = wdt_bf    + (long)DEPTHc * DINc * DTRc;
    __bf16* whead_bf  = wout_bf   + (long)DEPTHc * DIMc * DINc;   // 1,048,576

    // ---- one-shot bf16 conversions ----
    cvt_bf16_kernel<<<2048, 256, 0, stream>>>(series, series_bf, (long)Bc * NPc * Pc);
    seg_transpose_kernel<<<(NPc * DIMc * Pc) / 256, 256, 0, stream>>>(seg_w, wseg_bf);
    cvt_bf16_kernel<<<2048, 256, 0, stream>>>(in_proj_w, win_bf, (long)DEPTHc * 2 * DINc * DIMc);
    cvt_bf16_kernel<<<512, 256, 0, stream>>>(x_proj_w, wx_bf, (long)DEPTHc * 64 * DINc);
    cvt_bf16_kernel<<<256, 256, 0, stream>>>(dt_proj_w, wdt_bf, (long)DEPTHc * DINc * DTRc);
    cvt_bf16_kernel<<<2048, 256, 0, stream>>>(out_proj_w, wout_bf, (long)DEPTHc * DIMc * DINc);
    cvt_bf16_kernel<<<2048, 256, 0, stream>>>(head_w, whead_bf, (long)NCLSc * NPc * DIMc);

    // 1) seg linear: 64 batched GEMMs [128x128]*[128x512] -> x (f32)
    gemm_bf16_wmma_kernel<0, 4><<<dim3(DIMc / 64, Bc / 64, NPc), 128, 0, stream>>>(
        series_bf, NPc * Pc, Pc,
        wseg_bf, Pc, (long)DIMc * Pc,
        x, (long)NPc * DIMc, DIMc,
        seg_b, DIMc,
        nullptr, 0,
        Pc, 0);

    // 2) per-pathway LayerNorm (in place) + bf16 mirror for in_proj
    layernorm_kernel<<<ROWS, 256, 0, stream>>>(x, x, x_bf, ln_g, ln_b, DIMc, NPc);

    for (int i = 0; i < DEPTHc; ++i) {
        // 3) in_proj: [8192,512] x [512,2048] -> xz (f32)
        gemm_bf16_wmma_kernel<0, 4><<<dim3(2 * DINc / 64, ROWS / 64, 1), 128, 0, stream>>>(
            x_bf, DIMc, 0,
            win_bf + (long)i * 2 * DINc * DIMc, DIMc, 0,
            xz, 2 * DINc, 0,
            nullptr, 0,
            nullptr, 0,
            DIMc, 0);

        // 4) depthwise causal conv + SiLU -> uc (f32 + bf16)
        conv_silu_kernel<<<(ROWS * DINc) / 256, 256, 0, stream>>>(
            xz, conv_w + (long)i * DINc * DCONVc, conv_b + (long)i * DINc, uc, uc_bf);

        // 5) x_proj: [8192,1024] x [1024,64] -> xdbl (f32 + bf16)
        gemm_bf16_wmma_kernel<0, 4><<<dim3(1, ROWS / 64, 1), 128, 0, stream>>>(
            uc_bf, DINc, 0,
            wx_bf + (long)i * 64 * DINc, DINc, 0,
            xdbl, 64, 0,
            nullptr, 0,
            xdbl_bf, 64,
            DINc, 0);

        // 6) dt_proj (+bias) + softplus: [8192,32] x [32,1024] -> delta (f32)
        gemm_bf16_wmma_kernel<1, 4><<<dim3(DINc / 64, ROWS / 64, 1), 128, 0, stream>>>(
            xdbl_bf, 64, 0,
            wdt_bf + (long)i * DINc * DTRc, DTRc, 0,
            delta, DINc, 0,
            dt_proj_b + (long)i * DINc, 0,
            nullptr, 0,
            DTRc, 0);

        // 7) selective scan + D-skip + SiLU gate -> y (bf16 only)
        scan_kernel<<<dim3(DINc / 256, Bc), 256, 0, stream>>>(
            delta, uc, xz, xdbl,
            A_log + (long)i * DINc * DSTc, Dmat + (long)i * DINc, y_bf);

        // 8) out_proj: [8192,1024] x [1024,512] -> x (f32) + x_bf mirror
        gemm_bf16_wmma_kernel<0, 4><<<dim3(DIMc / 64, ROWS / 64, 1), 128, 0, stream>>>(
            y_bf, DINc, 0,
            wout_bf + (long)i * DIMc * DINc, DINc, 0,
            x, DIMc, 0,
            nullptr, 0,
            x_bf, DIMc,
            DINc, 0);
    }

    // 9) head LayerNorm over 32768 (x viewed [128, 32768]) -> x_bf (bf16)
    layernorm_kernel<<<Bc, 256, 0, stream>>>(x, x, x_bf, head_ln_g, head_ln_b,
                                             NPc * DIMc, 1);

    // 10) seed bias, split-K head GEMM with f32 atomics (K = 16 x 2048, NT=2)
    init_out_kernel<<<(Bc * NCLSc + 255) / 256, 256, 0, stream>>>(out, head_b);
    gemm_bf16_wmma_kernel<0, 2><<<dim3(1, Bc / 64, 16), 128, 0, stream>>>(
        x_bf, NPc * DIMc, 2048,
        whead_bf, NPc * DIMc, 2048,
        out, NCLSc, 0,
        nullptr, 0,
        nullptr, 0,
        2048, 1);

    (void)in_sizes; (void)n_in; (void)out_size; (void)ws_size;
}
